// EfficientNetB1OneLayerPatchCoreModel_56968446214308
// MI455X (gfx1250) — matrix operand order repaired and checked by reading, not compiled, not run on hardware
//
#include <hip/hip_runtime.h>
#include <hip/hip_bf16.h>
#include <stdint.h>

typedef __attribute__((ext_vector_type(8)))  _Float16 v8h;
typedef __attribute__((ext_vector_type(16))) _Float16 v16h;
typedef __attribute__((ext_vector_type(8)))  float    v8f;
typedef int v4i_g __attribute__((vector_size(16)));   // matches async-LDS builtin pointee

// Problem constants
constexpr int BATCH   = 8;
constexpr int C_IN    = 112;
constexpr int PATCH   = 256;        // 16*16
constexpr int Q_TOTAL = BATCH * PATCH;   // 2048
constexpr int DIM     = 128;
constexpr int N_BANK  = 200000;
constexpr int NN_K    = 3;
constexpr int TOPK_P  = 8;          // ceil(256*0.03)

// GEMM tiling
constexpr int NSLICE     = 100;
constexpr int SLICE_ROWS = 2000;             // 200000 / 100
constexpr int NCHUNK     = SLICE_ROWS / 16;  // 125
constexpr int LDSROW     = 136;              // 128 halves + 8 pad -> 272B row stride, conflict-free

// Workspace layout
constexpr size_t WS_BANK_OFF = 0;
constexpr size_t WS_BANK_SZ  = (size_t)N_BANK * DIM * 2;     // 51,200,000
constexpr size_t WS_EMB_OFF  = WS_BANK_OFF + WS_BANK_SZ;
constexpr size_t WS_EMB_SZ   = (size_t)Q_TOTAL * DIM * 2;    // 524,288
constexpr size_t WS_PART_OFF = WS_EMB_OFF + WS_EMB_SZ;       // 51,724,288 (16B aligned)

// ---- feature detection: async global->LDS copy (CDNA5) -------------------
#if defined(__HIP_DEVICE_COMPILE__) && defined(__has_builtin)
#  if __has_builtin(__builtin_amdgcn_global_load_async_to_lds_b128) && \
      __has_builtin(__builtin_amdgcn_s_wait_asynccnt)
#    define USE_ASYNC_LDS 1
#  endif
#endif
#ifndef USE_ASYNC_LDS
#  define USE_ASYNC_LDS 0
#endif

__device__ __forceinline__ uint32_t umn(uint32_t a, uint32_t b) { return a < b ? a : b; }
__device__ __forceinline__ uint32_t umx(uint32_t a, uint32_t b) { return a < b ? b : a; }

// Branchless sorted-triple insert on ordered-uint keys (biased-positive floats
// compare identically as u32). v_min_u32/v_max_u32: no canonicalization,
// VOP2 dual-issue eligible. Keeps t0 >= t1 >= t2.
__device__ __forceinline__ void insert3u(uint32_t& t0, uint32_t& t1, uint32_t& t2, uint32_t x) {
    const uint32_t m0 = umn(t0, x);
    t0 = umx(t0, x);
    const uint32_t m1 = umn(t1, m0);
    t1 = umx(t1, m0);
    t2 = umx(t2, m1);
}

// float top-3 insert for the small finalize kernel
__device__ __forceinline__ void insert3f(float& t0, float& t1, float& t2, float x) {
    const float m0 = fminf(t0, x);
    t0 = fmaxf(t0, x);
    const float m1 = fminf(t1, m0);
    t1 = fmaxf(t1, m0);
    t2 = fmaxf(t2, m1);
}

// -------------------------------------------------------------------------
// Kernel 1: patch embedding: NCHW gather, proj (no bias), L2-normalize, ->f16
// grid: 2048 blocks, 128 threads (thread = output dim d)
// -------------------------------------------------------------------------
__global__ __launch_bounds__(128) void embed_kernel(const float* __restrict__ fm,
                                                    const float* __restrict__ pw,
                                                    _Float16* __restrict__ embF16) {
    const int q   = blockIdx.x;          // 0..2047
    const int d   = threadIdx.x;         // 0..127
    const int b   = q >> 8;
    const int pix = q & 255;             // h*16+w
    const float* f = fm + (size_t)b * C_IN * PATCH + pix;
    const float* w = pw + (size_t)d * C_IN;
    float acc = 0.0f;
#pragma unroll 4
    for (int c = 0; c < C_IN; ++c) acc += f[(size_t)c * PATCH] * w[c];

    __shared__ float red[128];
    red[d] = acc * acc;
    __syncthreads();
    for (int s = 64; s > 0; s >>= 1) {
        if (d < s) red[d] += red[d + s];
        __syncthreads();
    }
    const float scale = 1.0f / fmaxf(sqrtf(red[0]), 1e-12f);
    embF16[(size_t)q * DIM + d] = (_Float16)(acc * scale);
}

// -------------------------------------------------------------------------
// Kernel 2: normalize memory bank rows, convert to f16
// grid: 200000 blocks, 128 threads
// -------------------------------------------------------------------------
__global__ __launch_bounds__(128) void bank_kernel(const float* __restrict__ bank,
                                                   _Float16* __restrict__ bankF16) {
    const int n = blockIdx.x;
    const int d = threadIdx.x;
    const float v = bank[(size_t)n * DIM + d];
    __shared__ float red[128];
    red[d] = v * v;
    __syncthreads();
    for (int s = 64; s > 0; s >>= 1) {
        if (d < s) red[d] += red[d + s];
        __syncthreads();
    }
    const float scale = 1.0f / fmaxf(sqrtf(red[0]), 1e-12f);
    bankF16[(size_t)n * DIM + d] = (_Float16)(v * scale);
}

// -------------------------------------------------------------------------
// Kernel 3: big GEMM (f16 WMMA, f32 acc) + running per-query top-3
// grid: (32 query-blocks of 64, 100 bank slices), block = 128 thr = 4 waves
// wave w owns queries [qblock*64 + w*16, +16); B tile shared via LDS
// -------------------------------------------------------------------------
__global__ __launch_bounds__(128) void gemm_topk_kernel(const _Float16* __restrict__ embF16,
                                                        const _Float16* __restrict__ bankF16,
                                                        float* __restrict__ partial) {
    __shared__ _Float16 tile[2][16 * LDSROW];   // double-buffered 16x128 f16 (+pad)

    const int tid   = threadIdx.x;
    const int lane  = tid & 31;
    const int wave  = tid >> 5;
    const int r     = lane & 15;    // A: row M, B: col N, C: col N
    const int hi    = lane >> 4;    // half-wave selector
    const int qbase = blockIdx.x * 64 + wave * 16;
    const int slice = blockIdx.y;
    const int rowBase0 = slice * SLICE_ROWS;

    // ---- load A fragments (16 queries x 128 K) into registers -----------
    v16h a[4];
    {
        const _Float16* arow = embF16 + (size_t)(qbase + r) * DIM;
#pragma unroll
        for (int kb = 0; kb < 4; ++kb) {
            const v8h lo  = *(const v8h*)(arow + kb * 32 + hi * 8);
            const v8h hi8 = *(const v8h*)(arow + kb * 32 + 16 + hi * 8);
            v16h t;
#pragma unroll
            for (int i = 0; i < 8; ++i) { t[i] = lo[i]; t[8 + i] = hi8[i]; }
            a[kb] = t;
        }
    }

    // Loop-invariant bias matrix: seed the WMMA chain with C = 2.0 so the
    // hardware produces sim+2 in [1,3] (strictly positive -> u32-orderable)
    // at zero per-chunk VALU cost.
    const v8f cbias = {2.f, 2.f, 2.f, 2.f, 2.f, 2.f, 2.f, 2.f};

    // per-lane running top-3 (u32 keys) for 8 queries
    uint32_t t0[8], t1[8], t2[8];
#pragma unroll
    for (int v = 0; v < 8; ++v) { t0[v] = 0u; t1[v] = 0u; t2[v] = 0u; }

    auto loadTile = [&](int buf, int rowBase) {
#pragma unroll
        for (int j = 0; j < 2; ++j) {
            const int idx = tid + j * 128;       // 256 x 16B covers 16x256B rows
            const int row = idx >> 4;
            const int c16 = idx & 15;
            const _Float16* g = bankF16 + (size_t)(rowBase + row) * DIM + c16 * 8;
            _Float16* l = &tile[buf][row * LDSROW + c16 * 8];
#if USE_ASYNC_LDS
            __builtin_amdgcn_global_load_async_to_lds_b128(
                (__attribute__((address_space(1))) v4i_g*)g,
                (__attribute__((address_space(3))) v4i_g*)l, 0, 0);
#else
            *(v8h*)l = *(const v8h*)g;
#endif
        }
    };

    loadTile(0, rowBase0);

    for (int c = 0; c < NCHUNK; ++c) {
        const int p = c & 1;
#if USE_ASYNC_LDS
        __builtin_amdgcn_s_wait_asynccnt(0);
#endif
        __syncthreads();   // chunk c visible to all waves; prev compute done
        if (c + 1 < NCHUNK) loadTile(1 - p, rowBase0 + (c + 1) * 16);

        // read ALL B fragments first (8x ds_load_b128), then issue the WMMAs,
        // so DS latency overlaps with matrix issue instead of serializing.
        const _Float16* brow = &tile[p][r * LDSROW + hi * 8];
        v16h bb[4];
#pragma unroll
        for (int kb = 0; kb < 4; ++kb) {
            const v8h b0 = *(const v8h*)(brow + kb * 32);
            const v8h b1 = *(const v8h*)(brow + kb * 32 + 16);
            v16h t;
#pragma unroll
            for (int i = 0; i < 8; ++i) { t[i] = b0[i]; t[8 + i] = b1[i]; }
            bb[kb] = t;
        }

        v8f acc = __builtin_amdgcn_wmma_f32_16x16x32_f16(
            false, a[0], false, bb[0], (short)0, cbias, false, false);
#pragma unroll
        for (int kb = 1; kb < 4; ++kb) {
            acc = __builtin_amdgcn_wmma_f32_16x16x32_f16(
                false, a[kb], false, bb[kb], (short)0, acc, false, false);
        }

        // fold biased sims into running top-3 via u32 min/max (branchless)
#pragma unroll
        for (int v = 0; v < 8; ++v)
            insert3u(t0[v], t1[v], t2[v], __float_as_uint(acc[v]));
    }

    // merge top-3 across the 16 column-lanes (same hi group) via xor shuffles
#pragma unroll
    for (int step = 1; step < 16; step <<= 1) {
#pragma unroll
        for (int v = 0; v < 8; ++v) {
            const uint32_t o0 = (uint32_t)__shfl_xor((int)t0[v], step, 32);
            const uint32_t o1 = (uint32_t)__shfl_xor((int)t1[v], step, 32);
            const uint32_t o2 = (uint32_t)__shfl_xor((int)t2[v], step, 32);
            insert3u(t0[v], t1[v], t2[v], o0);
            insert3u(t0[v], t1[v], t2[v], o1);
            insert3u(t0[v], t1[v], t2[v], o2);
        }
    }

    if (r == 0) {
#pragma unroll
        for (int v = 0; v < 8; ++v) {
            const int q = qbase + hi * 8 + v;
            float* o = partial + ((size_t)q * NSLICE + slice) * 3;
            o[0] = __uint_as_float(t0[v]) - 2.0f;   // un-bias back to sim
            o[1] = __uint_as_float(t1[v]) - 2.0f;
            o[2] = __uint_as_float(t2[v]) - 2.0f;
        }
    }
}

// -------------------------------------------------------------------------
// Kernel 4: merge per-slice top-3 -> distances -> per-image top-8 mean
// grid: 8 blocks (one per image), 256 threads (one per patch)
// -------------------------------------------------------------------------
__global__ __launch_bounds__(256) void finalize_kernel(const float* __restrict__ partial,
                                                       float* __restrict__ out) {
    const int b = blockIdx.x;
    const int p = threadIdx.x;
    const int q = b * PATCH + p;

    float t0 = -2.0f, t1 = -2.0f, t2 = -2.0f;
    const float* pp = partial + (size_t)q * NSLICE * 3;
    for (int s = 0; s < NSLICE; ++s) {
        insert3f(t0, t1, t2, pp[s * 3 + 0]);
        insert3f(t0, t1, t2, pp[s * 3 + 1]);
        insert3f(t0, t1, t2, pp[s * 3 + 2]);
    }
    const float dist = (sqrtf(fmaxf(2.0f - 2.0f * t0, 0.0f)) +
                        sqrtf(fmaxf(2.0f - 2.0f * t1, 0.0f)) +
                        sqrtf(fmaxf(2.0f - 2.0f * t2, 0.0f))) * (1.0f / NN_K);

    __shared__ float pd[PATCH];
    pd[p] = dist;
    __syncthreads();

    if (p == 0) {
        float top[TOPK_P];
#pragma unroll
        for (int i = 0; i < TOPK_P; ++i) top[i] = -1e30f;
        for (int j = 0; j < PATCH; ++j) {
            const float x = pd[j];
            if (x > top[TOPK_P - 1]) {
                top[TOPK_P - 1] = x;
#pragma unroll
                for (int k = TOPK_P - 1; k > 0; --k) {
                    if (top[k] > top[k - 1]) {
                        const float tmp = top[k - 1];
                        top[k - 1] = top[k];
                        top[k] = tmp;
                    }
                }
            }
        }
        float m = 0.0f;
#pragma unroll
        for (int i = 0; i < TOPK_P; ++i) m += top[i];
        out[b] = m * (1.0f / TOPK_P);
    }
}

extern "C" void kernel_launch(void* const* d_in, const int* in_sizes, int n_in,
                              void* d_out, int out_size, void* d_ws, size_t ws_size,
                              hipStream_t stream) {
    (void)in_sizes; (void)n_in; (void)out_size; (void)ws_size;
    const float* fm   = (const float*)d_in[0];   // [8,112,16,16]
    const float* pw   = (const float*)d_in[1];   // [128,112]
    const float* bank = (const float*)d_in[2];   // [200000,128]
    float* out = (float*)d_out;                  // [8]

    char* ws = (char*)d_ws;
    _Float16* bankF16 = (_Float16*)(ws + WS_BANK_OFF);
    _Float16* embF16  = (_Float16*)(ws + WS_EMB_OFF);
    float*    partial = (float*)(ws + WS_PART_OFF);

    embed_kernel<<<Q_TOTAL, 128, 0, stream>>>(fm, pw, embF16);
    bank_kernel<<<N_BANK, 128, 0, stream>>>(bank, bankF16);
    gemm_topk_kernel<<<dim3(Q_TOTAL / 64, NSLICE), 128, 0, stream>>>(embF16, bankF16, partial);
    finalize_kernel<<<BATCH, 256, 0, stream>>>(partial, out);
}
